// ObjectDetectionLoss_65352222375936
// MI455X (gfx1250) — matrix operand order repaired and checked by reading, not compiled, not run on hardware
//
#include <hip/hip_runtime.h>
#include <hip/hip_bf16.h>
#include <math.h>

// ---------------- problem constants ----------------
#define BB   4
#define NN   8400     // predictions
#define MM   32       // ground-truth boxes
#define CC   80       // classes
#define CP   96       // padded K for WMMA (3 x 32)
#define TILES_N (NN/16)   // 525 row tiles

#define EPSF 1e-7f
#define INVPI2_4 0.405284734569351f  // 4/pi^2

// ---------------- workspace layout (bytes, all 256B aligned) ----------------
#define OFF_DELTA  ((size_t)0)                         // B*N*96 f16  = 6,451,200
#define OFF_S      ((size_t)6451200)                   // B*N f32     =   134,400
#define OFF_PX     ((size_t)6585600)                   // B*N*4 f32   =   537,600
#define OFF_GX     ((size_t)7123200)                   // B*M*4 f32   =     2,048
#define OFF_GTBT   ((size_t)7125248)                   // B*32*96 f16 =    24,576
#define OFF_COSTT  ((size_t)7149824)                   // B*M*N f32   = 4,300,800
#define OFF_HUNG   ((size_t)11450624)                  // 4 * 169,216
#define HUNG_STRIDE ((size_t)169216)
#define OFF_MIDX   ((size_t)12127488)                  // B*M int
#define OFF_ACC    ((size_t)12128000)                  // accumulators

// ---------------- WMMA types ----------------
typedef _Float16 v16h __attribute__((ext_vector_type(16)));
typedef float    v8f  __attribute__((ext_vector_type(8)));
union H16 { v16h v; _Float16 h[16]; uint4 q[2]; };
union F8  { v8f  v; float f[8]; };

// ---------------- device math helpers ----------------
__device__ __forceinline__ float softplusf(float x) {
    // log(1 + e^x), stable
    return fmaxf(x, 0.0f) + log1pf(expf(-fabsf(x)));
}
__device__ __forceinline__ float sigmf(float x) { return 1.0f / (1.0f + expf(-x)); }
// focal(x, t=1) = 0.25 * softplus(-x) * (1-p)^2
__device__ __forceinline__ float focal_pos(float x) {
    float q = 1.0f - sigmf(x);
    return 0.25f * softplusf(-x) * q * q;
}
// focal(x, t=0) = 0.75 * softplus(x) * p^2
__device__ __forceinline__ float focal_neg(float x) {
    float p = sigmf(x);
    return 0.75f * softplusf(x) * p * p;
}
__device__ __forceinline__ float4 xywh2xyxy(const float* o) {
    float4 r;
    r.x = o[0] - 0.5f * o[2];
    r.y = o[1] - 0.5f * o[3];
    r.z = o[0] + 0.5f * o[2];
    r.w = o[1] + 0.5f * o[3];
    return r;
}
__device__ __forceinline__ float ciou4(float4 a, float4 b) {
    float w1 = a.z - a.x, h1 = a.w - a.y;
    float wg = b.z - b.x, hg = b.w - b.y;
    float iw = fmaxf(fminf(a.z, b.z) - fmaxf(a.x, b.x), 0.0f);
    float ih = fmaxf(fminf(a.w, b.w) - fmaxf(a.y, b.y), 0.0f);
    float inter = iw * ih;
    float uni   = w1 * h1 + wg * hg - inter;
    float iou   = inter / (uni + EPSF);
    float cw = fmaxf(a.z, b.z) - fminf(a.x, b.x);
    float ch = fmaxf(a.w, b.w) - fminf(a.y, b.y);
    float diag = cw * cw + ch * ch + EPSF;
    float dx = 0.5f * (a.x + a.z - b.x - b.z);
    float dy = 0.5f * (a.y + a.w - b.y - b.w);
    float rho2 = dx * dx + dy * dy;
    float diou = 1.0f - iou + rho2 / diag;
    float dv = atanf(wg / hg) - atanf(w1 / h1);
    float v  = INVPI2_4 * dv * dv;
    float alpha = v / (1.0f - iou + v + EPSF);
    return diou + alpha * v;
}

// ---------------- K0: zero accumulators ----------------
__global__ void k_init(float* acc) {
    if (threadIdx.x < 16) acc[threadIdx.x] = 0.0f;
}

// ---------------- K1: per-prediction prep ----------------
// S[b,n] = sum_c focal_neg(logit_c); delta[b,n,c] = focal_pos - focal_neg (f16, padded)
// px[b,n] = xyxy box
__global__ void k_prep_pred(const float* __restrict__ outputs,
                            _Float16* __restrict__ delta,
                            float* __restrict__ S,
                            float* __restrict__ px) {
    int idx = blockIdx.x * blockDim.x + threadIdx.x;
    if (idx >= BB * NN) return;
    const float* o = outputs + (size_t)idx * 85;
    *(float4*)(px + (size_t)idx * 4) = xywh2xyxy(o);
    _Float16* dr = delta + (size_t)idx * CP;
    float s = 0.0f;
    #pragma unroll 4
    for (int c = 0; c < CC; ++c) {
        float x  = o[5 + c];
        float fn = focal_neg(x);
        float fp = focal_pos(x);
        s += fn;
        dr[c] = (_Float16)(fp - fn);
    }
    for (int c = CC; c < CP; ++c) dr[c] = (_Float16)0.0f;
    S[idx] = s;
}

// ---------------- K1b: per-GT prep ----------------
// gtBT[b,m,k] = one-hot class (f16, padded); gx[b,m] = xyxy box
__global__ void k_prep_gt(const float* __restrict__ targets,
                          _Float16* __restrict__ gtBT,
                          float* __restrict__ gx) {
    int idx = blockIdx.x * blockDim.x + threadIdx.x;
    if (idx >= BB * MM) return;
    const float* t = targets + (size_t)idx * 85;
    *(float4*)(gx + (size_t)idx * 4) = xywh2xyxy(t);
    _Float16* r = gtBT + (size_t)idx * CP;
    for (int c = 0; c < CC; ++c) r[c] = (_Float16)t[5 + c];
    for (int c = CC; c < CP; ++c) r[c] = (_Float16)0.0f;
}

// ---------------- K2: WMMA class-cost GEMM fused with CIoU ----------------
// costT[b, m, n] = ciou(px[n], gx[m]) + (S[b,n] + delta[b,n,:]·onehot[m,:]) / 80
// One wave per 16-row tile: A = delta (16x32 f16 per step), B = gtBT^T (32x16),
// K = 96 -> 3 WMMA steps, 2 col tiles (M=32).
__global__ void __launch_bounds__(32)
k_wmma_cost(const _Float16* __restrict__ delta,
            const float* __restrict__ S,
            const float* __restrict__ px,
            const float* __restrict__ gx,
            const _Float16* __restrict__ gtBT,
            float* __restrict__ costT) {
    const int tile = blockIdx.x;      // 0..524
    const int b    = blockIdx.y;      // 0..3
    const int lane = threadIdx.x;     // 0..31
    const int lo   = lane & 15;
    const int g    = lane >> 4;

    // ---- A fragments: row = tile*16 + (lane%16); per ISA 16-bit A 16x32 layout:
    // halfs[0..7]  = A[row][kk*32 + g*8 + 0..7]
    // halfs[8..15] = A[row][kk*32 + 16 + g*8 + 0..7]
    const int arow = tile * 16 + lo;
    const _Float16* drow = delta + ((size_t)b * NN + arow) * CP;
    H16 afrag[3];
    #pragma unroll
    for (int kk = 0; kk < 3; ++kk) {
        afrag[kk].q[0] = *(const uint4*)(const void*)(drow + kk * 32 + g * 8);
        afrag[kk].q[1] = *(const uint4*)(const void*)(drow + kk * 32 + 16 + g * 8);
    }

    #pragma unroll
    for (int mt = 0; mt < 2; ++mt) {
        // ---- B fragment: col N' = lane%16, B[k][n] = gtBT[n][k];
        // halfs[0..7] = gtBT[mcol][kk*32 + g*8 + 0..7], halfs[8..15] = gtBT[mcol][kk*32 + 16 + g*8 + 0..7]
        const int mcol = mt * 16 + lo;
        const _Float16* brow = gtBT + ((size_t)b * MM + mcol) * CP;

        F8 acc;
        #pragma unroll
        for (int k = 0; k < 8; ++k) acc.f[k] = 0.0f;

        #pragma unroll
        for (int kk = 0; kk < 3; ++kk) {
            H16 bfrag;
            bfrag.q[0] = *(const uint4*)(const void*)(brow + kk * 32 + g * 8);
            bfrag.q[1] = *(const uint4*)(const void*)(brow + kk * 32 + 16 + g * 8);
            acc.v = __builtin_amdgcn_wmma_f32_16x16x32_f16(
                false, afrag[kk].v, false, bfrag.v,
                (short)0, acc.v, false, false);
        }

        // ---- fuse CIoU + write costT[b][mcol][n]
        const float4 gb = *(const float4*)(gx + ((size_t)b * MM + mcol) * 4);
        float* crow = costT + ((size_t)b * MM + mcol) * NN;
        #pragma unroll
        for (int r = 0; r < 8; ++r) {
            const int n = tile * 16 + g * 8 + r;   // D row = r + 8*(lane>=16)
            const float4 pb = *(const float4*)(px + ((size_t)b * NN + n) * 4);
            float cls = (S[(size_t)b * NN + n] + acc.f[r]) * (1.0f / (float)CC);
            crow[n] = ciou4(pb, gb) + cls;
        }
    }
}

// ---------------- K3: Hungarian (JV), one workgroup per batch ----------------
__global__ void __launch_bounds__(256)
k_hungarian(const float* __restrict__ costT, char* __restrict__ hws,
            int* __restrict__ midx) {
    const int b   = blockIdx.x;
    const int tid = threadIdx.x;
    const int bs  = blockDim.x;

    char* base = hws + (size_t)b * HUNG_STRIDE;
    float* v    = (float*)(base);
    float* minv = (float*)(base + 33792);
    int*   p    = (int*)  (base + 67584);
    int*   way  = (int*)  (base + 101376);
    int*   used = (int*)  (base + 135168);
    float* u    = (float*)(base + 168960);

    __shared__ float red_val[256];
    __shared__ int   red_idx[256];
    __shared__ float sh_delta;
    __shared__ int   sh_j0, sh_i0, sh_j1, sh_done;

    const float* Cb = costT + (size_t)b * MM * NN;
    const float INF = 1e30f;

    for (int j = tid; j <= NN; j += bs) { v[j] = 0.0f; p[j] = 0; }
    for (int i = tid; i <= MM; i += bs) u[i] = 0.0f;
    __syncthreads();

    for (int i = 1; i <= MM; ++i) {
        if (tid == 0) { p[0] = i; sh_j0 = 0; sh_done = 0; }
        for (int j = tid; j <= NN; j += bs) { minv[j] = INF; used[j] = 0; way[j] = 0; }
        __syncthreads();

        while (true) {
            if (tid == 0) { used[sh_j0] = 1; sh_i0 = p[sh_j0]; }
            __syncthreads();
            const int   j0  = sh_j0;
            const int   i0  = sh_i0;
            const float ui0 = u[i0];
            const float* Crow = Cb + (size_t)(i0 - 1) * NN;

            float lmin = INF; int lj = -1;
            for (int j = 1 + tid; j <= NN; j += bs) {
                if (!used[j]) {
                    float cur = Crow[j - 1] - ui0 - v[j];
                    if (cur < minv[j]) { minv[j] = cur; way[j] = j0; }
                    float mv = minv[j];
                    if (mv < lmin) { lmin = mv; lj = j; }
                }
            }
            red_val[tid] = lmin; red_idx[tid] = lj;
            __syncthreads();
            for (int s = bs >> 1; s > 0; s >>= 1) {
                if (tid < s && red_val[tid + s] < red_val[tid]) {
                    red_val[tid] = red_val[tid + s];
                    red_idx[tid] = red_idx[tid + s];
                }
                __syncthreads();
            }
            if (tid == 0) { sh_delta = red_val[0]; sh_j1 = red_idx[0]; }
            __syncthreads();
            const float delta = sh_delta;
            const int   j1    = sh_j1;
            if (j1 < 0) break;   // degenerate guard (uniform)

            for (int j = tid; j <= NN; j += bs) {
                if (used[j]) { u[p[j]] += delta; v[j] -= delta; }
                else          minv[j] -= delta;
            }
            __syncthreads();
            if (tid == 0) { sh_j0 = j1; sh_done = (p[j1] == 0); }
            __syncthreads();
            if (sh_done) break;
        }

        if (tid == 0) {   // augmenting path (serial, tiny)
            int j0 = sh_j0;
            while (j0 != 0) { int j1 = way[j0]; p[j0] = p[j1]; j0 = j1; }
        }
        __syncthreads();
    }

    for (int j = 1 + tid; j <= NN; j += bs)
        if (p[j] > 0) midx[b * MM + (p[j] - 1)] = j - 1;
}

// ---------------- K4a: objectness focal over all B*N ----------------
__global__ void __launch_bounds__(256)
k_obj(const float* __restrict__ outputs, const int* __restrict__ midx,
      float* __restrict__ acc) {
    __shared__ float red[256];
    int idx = blockIdx.x * blockDim.x + threadIdx.x;
    float val = 0.0f;
    if (idx < BB * NN) {
        int b = idx / NN, n = idx % NN;
        float x = outputs[(size_t)idx * 85 + 4];
        bool matched = false;
        #pragma unroll
        for (int i = 0; i < MM; ++i) matched |= (midx[b * MM + i] == n);
        val = matched ? focal_pos(x) : focal_neg(x);   // NEG_W = 1
    }
    red[threadIdx.x] = val;
    __syncthreads();
    for (int s = 128; s > 0; s >>= 1) {
        if (threadIdx.x < s) red[threadIdx.x] += red[threadIdx.x + s];
        __syncthreads();
    }
    if (threadIdx.x == 0) atomicAdd(acc, red[0]);
}

// ---------------- K4b: matched CIoU + class focal (f32) + final combine ----------------
__global__ void __launch_bounds__(128)
k_final(const float* __restrict__ outputs, const float* __restrict__ targets,
        const int* __restrict__ midx, const float* __restrict__ acc,
        float* __restrict__ out) {
    __shared__ float rs[128], rc[128];
    const int t = threadIdx.x;          // 0..127 = B*M pairs
    const int b = t / MM, i = t % MM;
    const int n = midx[b * MM + i];
    const float* o = outputs + ((size_t)b * NN + n) * 85;
    const float* g = targets + ((size_t)b * MM + i) * 85;
    float siou = ciou4(xywh2xyxy(o), xywh2xyxy(g));
    float cls = 0.0f;
    for (int c = 0; c < CC; ++c) {
        float x  = o[5 + c];
        float tc = g[5 + c];
        cls += tc * focal_pos(x) + (1.0f - tc) * focal_neg(x);
    }
    cls *= (1.0f / (float)CC);
    rs[t] = siou; rc[t] = cls;
    __syncthreads();
    for (int s = 64; s > 0; s >>= 1) {
        if (t < s) { rs[t] += rs[t + s]; rc[t] += rc[t + s]; }
        __syncthreads();
    }
    if (t == 0) {
        float m_siou = rs[0] * (1.0f / (float)(BB * MM));
        float m_cls  = rc[0] * (1.0f / (float)(BB * MM));
        float m_obj  = acc[0] * (1.0f / (float)(BB * NN));
        out[0] = 2.0f * m_siou + 2.0f * m_obj + 2.0f * m_cls;  // total
        out[1] = m_siou;
        out[2] = m_obj;
        out[3] = m_cls;
    }
}

// ---------------- host launcher ----------------
extern "C" void kernel_launch(void* const* d_in, const int* in_sizes, int n_in,
                              void* d_out, int out_size, void* d_ws, size_t ws_size,
                              hipStream_t stream) {
    const float* outputs = (const float*)d_in[0];
    const float* targets = (const float*)d_in[1];
    float* out = (float*)d_out;

    char* w = (char*)d_ws;
    _Float16* delta = (_Float16*)(w + OFF_DELTA);
    float*    S     = (float*)   (w + OFF_S);
    float*    px    = (float*)   (w + OFF_PX);
    float*    gx    = (float*)   (w + OFF_GX);
    _Float16* gtBT  = (_Float16*)(w + OFF_GTBT);
    float*    costT = (float*)   (w + OFF_COSTT);
    char*     hws   =            (w + OFF_HUNG);
    int*      midx  = (int*)     (w + OFF_MIDX);
    float*    acc   = (float*)   (w + OFF_ACC);

    k_init<<<1, 32, 0, stream>>>(acc);
    k_prep_pred<<<(BB * NN + 255) / 256, 256, 0, stream>>>(outputs, delta, S, px);
    k_prep_gt<<<1, 128, 0, stream>>>(targets, gtBT, gx);
    k_wmma_cost<<<dim3(TILES_N, BB), 32, 0, stream>>>(delta, S, px, gx, gtBT, costT);
    k_hungarian<<<BB, 256, 0, stream>>>(costT, hws, midx);
    k_obj<<<(BB * NN + 255) / 256, 256, 0, stream>>>(outputs, midx, acc);
    k_final<<<1, 128, 0, stream>>>(outputs, targets, midx, acc, out);
}
